// SlotAttention_7954279432755
// MI455X (gfx1250) — matrix operand order repaired and checked by reading, not compiled, not run on hardware
//
#include <hip/hip_runtime.h>

// Slot Attention for MI455X (gfx1250): bf16 WMMA for the big GEMMs,
// TDM (tensor_load_to_lds) staging of shared operands into LDS,
// dual WMMA accumulator chains per wave to hide XDL latency.
// Slots live in d_out and are updated in place.

#define B_   64
#define N_   4096
#define IND_ 256
#define S_   8
#define D_   256
#define SP_  16   // S padded to 16 for WMMA M dimension

typedef __attribute__((ext_vector_type(16))) __bf16 v16bf;
typedef __attribute__((ext_vector_type(8)))  __bf16 v8bf;
typedef __attribute__((ext_vector_type(8)))  float  v8f;
typedef __attribute__((ext_vector_type(4)))  float  v4f;
typedef __attribute__((ext_vector_type(4)))  unsigned int v4u;
typedef __attribute__((ext_vector_type(8)))  int    v8i;
typedef __attribute__((ext_vector_type(4)))  int    v4i;

static __device__ __forceinline__ __bf16 cvt_bf16(float f) {
  unsigned u = __builtin_bit_cast(unsigned, f);
  u += 0x7FFFu + ((u >> 16) & 1u);          // round-to-nearest-even
  unsigned short s = (unsigned short)(u >> 16);
  return __builtin_bit_cast(__bf16, s);
}

static __device__ __forceinline__ v16bf join8(v8bf lo, v8bf hi) {
  v16bf r;
#pragma unroll
  for (int i = 0; i < 8; ++i) { r[i] = lo[i]; r[i + 8] = hi[i]; }
  return r;
}

static __device__ __forceinline__ v8f wmma_bf16(v16bf a, v16bf b, v8f c) {
  return __builtin_amdgcn_wmma_f32_16x16x32_bf16(
      false, a, false, b, (short)0, c, false, false);
}

// ---- Tensor Data Mover: 2-D tile (bf16) global -> LDS, per ISA D# layout --
static __device__ __forceinline__ void tdm_load_2d_bf16(
    unsigned lds_addr, const void* gptr, unsigned tensor_d0,
    unsigned tensor_d1, unsigned tile_d0, unsigned tile_d1,
    unsigned long long stride0 /* elements */) {
  unsigned long long ga = (unsigned long long)gptr;
  v4u g0;
  g0[0] = 1u;                                        // count=1 (user D#)
  g0[1] = lds_addr;                                  // LDS byte address
  g0[2] = (unsigned)(ga & 0xFFFFFFFFu);              // global_addr lo
  g0[3] = (unsigned)((ga >> 32) & 0x01FFFFFFu) | (2u << 30);  // type=2
  v8i g1;
  g1[0] = (int)(1u << 16);                           // wg_mask=0, data_size=2B
  g1[1] = (int)((tensor_d0 & 0xFFFFu) << 16);        // tensor_dim0[15:0]
  g1[2] = (int)((tensor_d0 >> 16) | ((tensor_d1 & 0xFFFFu) << 16));
  g1[3] = (int)((tensor_d1 >> 16) | (tile_d0 << 16));// tile_dim0
  g1[4] = (int)(tile_d1 & 0xFFFFu);                  // tile_dim1, tile_dim2=0
  g1[5] = (int)(stride0 & 0xFFFFFFFFu);              // tensor_dim0_stride lo
  g1[6] = (int)((stride0 >> 32) & 0xFFFFu);          // stride hi, stride1=0
  g1[7] = 0;
  v4i z4 = {};
#if __clang_major__ >= 23
  v8i z8 = {};
  __builtin_amdgcn_tensor_load_to_lds(g0, g1, z4, z4, z8, 0);
#else
  __builtin_amdgcn_tensor_load_to_lds(g0, g1, z4, z4, 0);
#endif
}

// ---------------------------------------------------------------- prep ----
__global__ __launch_bounds__(256) void k_cvt(const float* __restrict__ s,
                                             __bf16* __restrict__ d, int n) {
  int i = blockIdx.x * 256 + threadIdx.x;
  if (i < n) d[i] = cvt_bf16(s[i]);
}

__global__ __launch_bounds__(256) void k_init_slots(
    const float* __restrict__ mu, const float* __restrict__ sg,
    const float* __restrict__ nz, float* __restrict__ out) {
  int i = blockIdx.x * 256 + threadIdx.x;            // 131072 elements
  out[i] = mu[i & (S_ * D_ - 1)] + sg[i & (S_ * D_ - 1)] * nz[i];
}

__global__ __launch_bounds__(256) void k_zero(unsigned* __restrict__ p, int n) {
  int i = blockIdx.x * 256 + threadIdx.x;
  if (i < n) p[i] = 0u;
}

// -------------------------------------------------- K/V projection (WMMA) --
// K stored [B*N, D] bf16 (row-major); V stored transposed [B, D, N] bf16.
// Weight tiles staged in LDS once per block; dual acc chains (K & V).
__global__ __launch_bounds__(256) void k_kv_proj(
    const float* __restrict__ inputs, const __bf16* __restrict__ wkb,
    const __bf16* __restrict__ wvb, const float* __restrict__ bk,
    const float* __restrict__ bv, __bf16* __restrict__ Kbf,
    __bf16* __restrict__ Vt) {
  __shared__ __bf16 sW[2][16][256];                  // 16 KB weight tile (K,V)
  const int wave = threadIdx.x >> 5;
  const int lane = threadIdx.x & 31;
  const int l15 = lane & 15;
  const int hi = lane >> 4;
  const long rowBase = (long)blockIdx.x * 128 + wave * 16;
  const long g = rowBase + l15;
  const float* arow = inputs + g * IND_;

  // Load the wave's 16x256 input stripe, convert to bf16 A-fragments.
  v16bf A[8];
#pragma unroll
  for (int k8 = 0; k8 < 8; ++k8) {
    int kb = k8 * 32 + hi * 16;
    v4f x0 = *(const v4f*)(arow + kb);
    v4f x1 = *(const v4f*)(arow + kb + 4);
    v4f x2 = *(const v4f*)(arow + kb + 8);
    v4f x3 = *(const v4f*)(arow + kb + 12);
    v16bf a;
#pragma unroll
    for (int i = 0; i < 4; ++i) {
      a[i]      = cvt_bf16(x0[i]);
      a[i + 4]  = cvt_bf16(x1[i]);
      a[i + 8]  = cvt_bf16(x2[i]);
      a[i + 12] = cvt_bf16(x3[i]);
    }
    A[k8] = a;
  }

  const long bIdx = rowBase >> 12;        // batch   (128 | 4096)
  const long nInB = rowBase & (N_ - 1);   // n of stripe row 0

  for (int ct = 0; ct < 16; ++ct) {
    __syncthreads();                       // previous tile's LDS reads done
    // Cooperative stage of Wk/Wv 16x256 tiles: 1024 b128 chunks / 256 thr.
#pragma unroll
    for (int j = 0; j < 4; ++j) {
      int cid = threadIdx.x + j * 256;     // 0..1023
      int w = cid >> 9, rem = cid & 511;
      int row = rem >> 5, col8 = (rem & 31) * 8;
      const __bf16* src = (w ? wvb : wkb) + ((long)(ct * 16 + row)) * IND_ + col8;
      *(v8bf*)&sW[w][row][col8] = *(const v8bf*)src;
    }
    __syncthreads();

    int d = ct * 16 + l15;
    v8f accK = {};
    v8f accV = {};
#pragma unroll
    for (int k8 = 0; k8 < 8; ++k8) {
      int kk = k8 * 32 + hi * 16;
      v16bf Bk = join8(*(const v8bf*)&sW[0][l15][kk],
                       *(const v8bf*)&sW[0][l15][kk + 8]);
      v16bf Bv = join8(*(const v8bf*)&sW[1][l15][kk],
                       *(const v8bf*)&sW[1][l15][kk + 8]);
      accK = wmma_bf16(A[k8], Bk, accK);
      accV = wmma_bf16(A[k8], Bv, accV);
    }
    float bko = bk[d], bvo = bv[d];
    // K[n, d]: lane owns column d, 8 strided rows
#pragma unroll
    for (int i = 0; i < 8; ++i) {
      long r = rowBase + hi * 8 + i;
      Kbf[r * D_ + d] = cvt_bf16(accK[i] + bko);
    }
    // Vt[b, d, n]: 8 contiguous n per lane -> one b128 store
    long nb = nInB + hi * 8;
    v8bf pk;
#pragma unroll
    for (int i = 0; i < 8; ++i) pk[i] = cvt_bf16(accV[i] + bvo);
    *(v8bf*)(Vt + ((bIdx * D_ + d) * (long)N_) + nb) = pk;
  }
}

// ------------------------------------------- LayerNorm(slots) @ Wq^T + bq --
__global__ __launch_bounds__(256) void k_qproj(
    const float* __restrict__ slots, const float* __restrict__ lng,
    const float* __restrict__ lnb, const float* __restrict__ Wq,
    const float* __restrict__ bq, __bf16* __restrict__ qpad) {
  __shared__ float sxn[S_][D_];
  const int b = blockIdx.x;
  const int wave = threadIdx.x >> 5;      // slot index
  const int lane = threadIdx.x & 31;
  const float* row = slots + ((long)b * S_ + wave) * D_;
  float x[8];
  float sum = 0.f;
#pragma unroll
  for (int i = 0; i < 8; ++i) { x[i] = row[lane * 8 + i]; sum += x[i]; }
#pragma unroll
  for (int m = 16; m; m >>= 1) sum += __shfl_xor(sum, m, 32);
  float mu = sum * (1.0f / D_);
  float vs = 0.f;
#pragma unroll
  for (int i = 0; i < 8; ++i) { float t = x[i] - mu; vs += t * t; }
#pragma unroll
  for (int m = 16; m; m >>= 1) vs += __shfl_xor(vs, m, 32);
  float rs = rsqrtf(vs * (1.0f / D_) + 1e-5f);
#pragma unroll
  for (int i = 0; i < 8; ++i) {
    int d = lane * 8 + i;
    sxn[wave][d] = (x[i] - mu) * rs * lng[d] + lnb[d];
  }
  __syncthreads();
  const int s = wave;
  const int dbase = lane * 8;
  for (int i = 0; i < 8; ++i) {
    int dout = dbase + i;
    const float* wr = Wq + (long)dout * D_;
    float acc = bq[dout];
    for (int kk = 0; kk < D_; ++kk) acc += sxn[s][kk] * wr[kk];
    qpad[((long)b * SP_ + s) * D_ + dout] = cvt_bf16(acc);
  }
}

// -------------------------------------------------- attn logits = q K^T ----
// q staged in LDS (shared A across waves); dual n-tile acc chains per wave.
__global__ __launch_bounds__(256) void k_logits(
    const __bf16* __restrict__ qpad, const __bf16* __restrict__ Kbf,
    float* __restrict__ logits) {
  __shared__ __bf16 sQ[16][256];          // 8 KB
  const int b = blockIdx.x >> 4;
  const int grp = blockIdx.x & 15;
  const int wave = threadIdx.x >> 5;
  const int lane = threadIdx.x & 31;
  const int l15 = lane & 15, hi = lane >> 4;

  // Cooperative stage of qpad[b] (16x256 bf16): 512 chunks / 256 threads.
#pragma unroll
  for (int j = 0; j < 2; ++j) {
    int cid = threadIdx.x + j * 256;
    int row = cid >> 5, col8 = (cid & 31) * 8;
    *(v8bf*)&sQ[row][col8] =
        *(const v8bf*)(qpad + (long)b * SP_ * D_ + (long)row * D_ + col8);
  }
  __syncthreads();

  v16bf A[8];
#pragma unroll
  for (int k8 = 0; k8 < 8; ++k8) {
    int kb = k8 * 32 + hi * 16;
    A[k8] = join8(*(const v8bf*)&sQ[l15][kb], *(const v8bf*)&sQ[l15][kb + 8]);
  }
  const int n0 = (grp * 16 + wave) * 16 + l15;
  const int n1 = n0 + 128;
  const __bf16* kr0 = Kbf + ((long)b * N_ + n0) * D_;
  const __bf16* kr1 = Kbf + ((long)b * N_ + n1) * D_;
  v8f acc0 = {};
  v8f acc1 = {};
#pragma unroll
  for (int k8 = 0; k8 < 8; ++k8) {
    int kk = k8 * 32 + hi * 16;
    v16bf B0 = join8(*(const v8bf*)(kr0 + kk), *(const v8bf*)(kr0 + kk + 8));
    v16bf B1 = join8(*(const v8bf*)(kr1 + kk), *(const v8bf*)(kr1 + kk + 8));
    acc0 = wmma_bf16(A[k8], B0, acc0);
    acc1 = wmma_bf16(A[k8], B1, acc1);
  }
  const float sc = 0.0625f;  // 1/sqrt(256)
  if (hi == 0) {
#pragma unroll
    for (int i = 0; i < 8; ++i) {
      logits[((long)b * S_ + i) * N_ + n0] = acc0[i] * sc;
      logits[((long)b * S_ + i) * N_ + n1] = acc1[i] * sc;
    }
  }
}

// ------------------------------------------------------ softmax over N ----
__global__ __launch_bounds__(256) void k_softmax(
    const float* __restrict__ logits, __bf16* __restrict__ attnp) {
  const int b = blockIdx.x >> 4;
  const int s = blockIdx.x & 15;
  __bf16* arow = attnp + ((long)b * SP_ + s) * N_;
  const int t = threadIdx.x;
  if (s >= S_) {                          // padded rows stay zero
    unsigned* z = (unsigned*)arow;
    for (int i = t; i < N_ / 2; i += 256) z[i] = 0u;
    return;
  }
  const float* lrow = logits + ((long)b * S_ + s) * N_;
  __shared__ float red[256];
  float v[16];
  float mx = -3.0e38f;
#pragma unroll
  for (int i = 0; i < 16; ++i) {
    v[i] = lrow[t + i * 256];
    mx = fmaxf(mx, v[i]);
  }
  red[t] = mx;
  __syncthreads();
  for (int o = 128; o; o >>= 1) {
    if (t < o) red[t] = fmaxf(red[t], red[t + o]);
    __syncthreads();
  }
  mx = red[0];
  __syncthreads();
  float sum = 0.f;
#pragma unroll
  for (int i = 0; i < 16; ++i) { v[i] = __expf(v[i] - mx); sum += v[i]; }
  red[t] = sum;
  __syncthreads();
  for (int o = 128; o; o >>= 1) {
    if (t < o) red[t] += red[t + o];
    __syncthreads();
  }
  float inv = 1.0f / red[0];
#pragma unroll
  for (int i = 0; i < 16; ++i) arow[t + i * 256] = cvt_bf16(v[i] * inv);
}

// ------------------------------------------------- updates = attn @ v -----
// attn chunk staged into LDS by the Tensor Data Mover (32 KB x 4 chunks);
// each wave runs two independent d-tile WMMA chains sharing the LDS A.
__global__ __launch_bounds__(256) void k_updates(
    const __bf16* __restrict__ attnp, const __bf16* __restrict__ Vt,
    float* __restrict__ upd) {
  __shared__ __bf16 sA[16][1024];          // 32 KB attn chunk [s][n]
  const int b = blockIdx.x;
  const int wave = threadIdx.x >> 5;
  const int lane = threadIdx.x & 31;
  const int l15 = lane & 15, hi = lane >> 4;
  const int d0 = wave * 16 + l15;          // d tiles 0..7
  const int d1 = 128 + wave * 16 + l15;    // d tiles 8..15
  const __bf16* vr0 = Vt + ((long)b * D_ + d0) * (long)N_;
  const __bf16* vr1 = Vt + ((long)b * D_ + d1) * (long)N_;
  v8f acc0 = {};
  v8f acc1 = {};
  for (int chunk = 0; chunk < 4; ++chunk) {
    __syncthreads();                       // previous chunk's reads done
    if (wave == 0) {
      unsigned lds = (unsigned)(unsigned long long)(const void*)&sA[0][0];
      const __bf16* gsrc = attnp + (long)b * SP_ * N_ + chunk * 1024;
      tdm_load_2d_bf16(lds, gsrc, /*tensor_d0=*/N_, /*tensor_d1=*/SP_,
                       /*tile_d0=*/1024, /*tile_d1=*/SP_, /*stride0=*/N_);
      __builtin_amdgcn_s_wait_tensorcnt(0);
    }
    __syncthreads();                       // LDS tile visible to all waves
    const int base = chunk * 32;
    for (int k8 = 0; k8 < 32; ++k8) {
      int ka = k8 * 32 + hi * 16;
      v16bf A = join8(*(const v8bf*)&sA[l15][ka],
                      *(const v8bf*)&sA[l15][ka + 8]);
      int kb = (base + k8) * 32 + hi * 16;
      v16bf B0 = join8(*(const v8bf*)(vr0 + kb), *(const v8bf*)(vr0 + kb + 8));
      v16bf B1 = join8(*(const v8bf*)(vr1 + kb), *(const v8bf*)(vr1 + kb + 8));
      __builtin_prefetch(vr0 + kb + 1024, 0, 1);   // global_prefetch ahead
      acc0 = wmma_bf16(A, B0, acc0);
      acc1 = wmma_bf16(A, B1, acc1);
    }
  }
  if (hi == 0) {
#pragma unroll
    for (int i = 0; i < 8; ++i) {
      upd[((long)b * S_ + i) * D_ + d0] = acc0[i];
      upd[((long)b * S_ + i) * D_ + d1] = acc1[i];
    }
  }
}

// ----------------------------- GRU cell + residual (slots updated in place)
__global__ __launch_bounds__(256) void k_gru(
    const float* __restrict__ upd, const float* __restrict__ Wih,
    const float* __restrict__ bih, const float* __restrict__ Whh,
    const float* __restrict__ bhh, float* __restrict__ slots) {
  __shared__ float sh[S_][D_];
  __shared__ float sx[S_][D_];
  const int b = blockIdx.x;
  const int t = threadIdx.x;
  for (int i = t; i < S_ * D_; i += 256) {
    sh[0][i] = slots[(long)b * S_ * D_ + i];
    sx[0][i] = upd[(long)b * S_ * D_ + i];
  }
  __syncthreads();
  const int d = t;
  const float* wr_r = Wih + (long)d * D_;
  const float* wr_z = Wih + (long)(D_ + d) * D_;
  const float* wr_n = Wih + (long)(2 * D_ + d) * D_;
  const float* wh_r = Whh + (long)d * D_;
  const float* wh_z = Whh + (long)(D_ + d) * D_;
  const float* wh_n = Whh + (long)(2 * D_ + d) * D_;
  for (int s = 0; s < S_; ++s) {
    float ir = bih[d], iz = bih[D_ + d], in_ = bih[2 * D_ + d];
    float hr = bhh[d], hz = bhh[D_ + d], hn = bhh[2 * D_ + d];
    for (int kk = 0; kk < D_; ++kk) {
      float xv = sx[s][kk], hv = sh[s][kk];
      ir += xv * wr_r[kk]; iz += xv * wr_z[kk]; in_ += xv * wr_n[kk];
      hr += hv * wh_r[kk]; hz += hv * wh_z[kk]; hn += hv * wh_n[kk];
    }
    float r = 1.f / (1.f + __expf(-(ir + hr)));
    float z = 1.f / (1.f + __expf(-(iz + hz)));
    float nn = tanhf(in_ + r * hn);
    float h = sh[s][d];
    slots[((long)b * S_ + s) * D_ + d] = h + ((1.f - z) * nn + z * h);
  }
}

// -------------------------- hidden = relu(LN(slots) @ W1^T + b1) ----------
__global__ __launch_bounds__(256) void k_mlp1(
    const float* __restrict__ slots, const float* __restrict__ lng,
    const float* __restrict__ lnb, const float* __restrict__ W1,
    const float* __restrict__ b1, float* __restrict__ hidden) {
  __shared__ float sxn[S_][D_];
  const int b = blockIdx.x;
  const int wave = threadIdx.x >> 5;
  const int lane = threadIdx.x & 31;
  const float* row = slots + ((long)b * S_ + wave) * D_;
  float x[8];
  float sum = 0.f;
#pragma unroll
  for (int i = 0; i < 8; ++i) { x[i] = row[lane * 8 + i]; sum += x[i]; }
#pragma unroll
  for (int m = 16; m; m >>= 1) sum += __shfl_xor(sum, m, 32);
  float mu = sum * (1.0f / D_);
  float vs = 0.f;
#pragma unroll
  for (int i = 0; i < 8; ++i) { float t = x[i] - mu; vs += t * t; }
#pragma unroll
  for (int m = 16; m; m >>= 1) vs += __shfl_xor(vs, m, 32);
  float rs = rsqrtf(vs * (1.0f / D_) + 1e-5f);
#pragma unroll
  for (int i = 0; i < 8; ++i) {
    int d = lane * 8 + i;
    sxn[wave][d] = (x[i] - mu) * rs * lng[d] + lnb[d];
  }
  __syncthreads();
  const int s = wave;
  const int dbase = lane * 8;
  for (int i = 0; i < 8; ++i) {
    int dout = dbase + i;
    const float* wr = W1 + (long)dout * D_;
    float acc = b1[dout];
    for (int kk = 0; kk < D_; ++kk) acc += sxn[s][kk] * wr[kk];
    hidden[((long)b * S_ + s) * D_ + dout] = fmaxf(acc, 0.f);
  }
}

// -------------------------------- slots += hidden @ W2^T + b2 -------------
__global__ __launch_bounds__(256) void k_mlp2(
    const float* __restrict__ hidden, const float* __restrict__ W2,
    const float* __restrict__ b2, float* __restrict__ slots) {
  __shared__ float shd[S_][D_];
  const int b = blockIdx.x;
  const int t = threadIdx.x;
  for (int i = t; i < S_ * D_; i += 256)
    shd[0][i] = hidden[(long)b * S_ * D_ + i];
  __syncthreads();
  const int d = t;
  const float* wr = W2 + (long)d * D_;
  for (int s = 0; s < S_; ++s) {
    float acc = b2[d];
    for (int kk = 0; kk < D_; ++kk) acc += shd[s][kk] * wr[kk];
    slots[((long)b * S_ + s) * D_ + d] += acc;
  }
}

// ------------------------------------------------------------- host -------
extern "C" void kernel_launch(void* const* d_in, const int* in_sizes, int n_in,
                              void* d_out, int out_size, void* d_ws,
                              size_t ws_size, hipStream_t stream) {
  (void)in_sizes; (void)n_in; (void)out_size; (void)ws_size;
  const float* inputs    = (const float*)d_in[0];
  const float* noise     = (const float*)d_in[1];
  const float* slots_mu  = (const float*)d_in[2];
  const float* slots_sg  = (const float*)d_in[3];
  const float* Wq        = (const float*)d_in[4];
  const float* bq        = (const float*)d_in[5];
  const float* Wk        = (const float*)d_in[6];
  const float* bk        = (const float*)d_in[7];
  const float* Wv        = (const float*)d_in[8];
  const float* bv        = (const float*)d_in[9];
  const float* Wih       = (const float*)d_in[10];
  const float* bih       = (const float*)d_in[11];
  const float* Whh       = (const float*)d_in[12];
  const float* bhh       = (const float*)d_in[13];
  const float* W1        = (const float*)d_in[14];
  const float* b1        = (const float*)d_in[15];
  const float* W2        = (const float*)d_in[16];
  const float* b2        = (const float*)d_in[17];
  const float* ln_s_g    = (const float*)d_in[18];
  const float* ln_s_b    = (const float*)d_in[19];
  const float* ln_m_g    = (const float*)d_in[20];
  const float* ln_m_b    = (const float*)d_in[21];
  float* slots = (float*)d_out;   // [B, S, D] updated in place

  char* ws = (char*)d_ws;
  size_t off = 0;
  auto alloc = [&](size_t bytes) -> void* {
    off = (off + 255) & ~(size_t)255;
    void* p = ws + off;
    off += bytes;
    return p;
  };
  __bf16* Kbf    = (__bf16*)alloc((size_t)B_ * N_ * D_ * 2);   // [B*N, D]
  __bf16* Vt     = (__bf16*)alloc((size_t)B_ * D_ * N_ * 2);   // [B, D, N]
  __bf16* qpad   = (__bf16*)alloc((size_t)B_ * SP_ * D_ * 2);  // [B, 16, D]
  __bf16* attnp  = (__bf16*)alloc((size_t)B_ * SP_ * N_ * 2);  // [B, 16, N]
  float*  logits = (float*)alloc((size_t)B_ * S_ * N_ * 4);
  float*  upd    = (float*)alloc((size_t)B_ * S_ * D_ * 4);
  float*  hidden = (float*)alloc((size_t)B_ * S_ * D_ * 4);
  __bf16* wkb    = (__bf16*)alloc((size_t)D_ * IND_ * 2);
  __bf16* wvb    = (__bf16*)alloc((size_t)D_ * IND_ * 2);

  // prep: bf16 weights, slot init, zero padded q rows
  k_cvt<<<(D_ * IND_ + 255) / 256, 256, 0, stream>>>(Wk, wkb, D_ * IND_);
  k_cvt<<<(D_ * IND_ + 255) / 256, 256, 0, stream>>>(Wv, wvb, D_ * IND_);
  k_init_slots<<<(B_ * S_ * D_) / 256, 256, 0, stream>>>(slots_mu, slots_sg,
                                                         noise, slots);
  k_zero<<<(B_ * SP_ * D_ / 2 + 255) / 256, 256, 0, stream>>>(
      (unsigned*)qpad, B_ * SP_ * D_ / 2);

  // k/v projection (loop-invariant, hoisted)
  k_kv_proj<<<(B_ * N_) / 128, 256, 0, stream>>>(inputs, wkb, wvb, bk, bv,
                                                 Kbf, Vt);

  for (int it = 0; it < 3; ++it) {
    k_qproj<<<B_, 256, 0, stream>>>(slots, ln_s_g, ln_s_b, Wq, bq, qpad);
    k_logits<<<B_ * 16, 256, 0, stream>>>(qpad, Kbf, logits);
    k_softmax<<<B_ * SP_, 256, 0, stream>>>(logits, attnp);
    k_updates<<<B_, 256, 0, stream>>>(attnp, Vt, upd);
    k_gru<<<B_, 256, 0, stream>>>(upd, Wih, bih, Whh, bhh, slots);
    k_mlp1<<<B_, 256, 0, stream>>>(slots, ln_m_g, ln_m_b, W1, b1, hidden);
    k_mlp2<<<B_, 256, 0, stream>>>(hidden, W2, b2, slots);
  }
}
